// Transformer_15668040696380
// MI455X (gfx1250) — matrix operand order repaired and checked by reading, not compiled, not run on hardware
//
#include <hip/hip_runtime.h>
#include <hip/hip_bf16.h>

typedef __bf16 bf16;
typedef __attribute__((ext_vector_type(16))) __bf16 v16bf;
typedef __attribute__((ext_vector_type(8)))  float  v8f;

#define D_MODEL  1024
#define SEQ      2048
#define BATCHN   2
#define NTOK     (BATCHN * SEQ)     // 4096 rows in every GEMM
#define D_MLP    4096
#define D_HEAD   64
#define N_HEADS  16
#define N_LAYERS 4
#define D_VOCAB  32000
#define ATT_SCALE 0.125f            // 1/sqrt(64)

// ---------------------------------------------------------------------------
// WMMA fragment helpers (CDNA5 16x16x32 bf16, wave32; layouts per ISA 7.12.2)
// ---------------------------------------------------------------------------

__device__ __forceinline__ v8f wmma_bf16(v16bf a, v16bf b, v8f c) {
  // D = A(16x32) * B(32x16) + C(16x16), f32 accumulate
  return __builtin_amdgcn_wmma_f32_16x16x32_bf16(
      /*neg_a=*/false, a, /*neg_b=*/false, b,
      /*c_mod=*/(short)0, c, /*reuse_a=*/false, /*reuse_b=*/false);
}

// A-matrix 16x32 bf16 from row-major [M,K]:
// lane<16: row=lane,     elems 0..7 = K k0+0..7,  elems 8..15 = K k0+16..23
// lane>=16: row=lane-16, elems 0..7 = K k0+8..15, elems 8..15 = K k0+24..31
__device__ __forceinline__ v16bf load_a_bf16(const bf16* __restrict__ A, int ldk,
                                             int m0, int k0) {
  int lane = threadIdx.x & 31;
  int half = lane >> 4, r = lane & 15;
  const bf16* p = A + (size_t)(m0 + r) * ldk + k0 + half * 8;
  union { v16bf v; float4 q[2]; } u;
  u.q[0] = *reinterpret_cast<const float4*>(p);
  u.q[1] = *reinterpret_cast<const float4*>(p + 16);
  return u.v;
}

// B-matrix 32x16: column n per lane (n = lane&15), elems 0..7 = K 8*half+0..7, elems 8..15 = K 16+8*half+0..7.
// Source rows of length K are contiguous (bf16 weights stored [N,K], k-tiles).
__device__ __forceinline__ v16bf load_b_bf16_rows(const bf16* __restrict__ Bm, int ldk,
                                                  int n0, int k0) {
  int lane = threadIdx.x & 31;
  int half = lane >> 4, c = lane & 15;
  const bf16* p = Bm + (size_t)(n0 + c) * ldk + k0 + half * 8;
  union { v16bf v; float4 q[2]; } u;
  u.q[0] = *reinterpret_cast<const float4*>(p);
  u.q[1] = *reinterpret_cast<const float4*>(p + 16);
  return u.v;
}

// B from [K,N]-stored bf16 (V tiles: [keys, 64]): column n strided by ldn.
__device__ __forceinline__ v16bf load_b_bf16_cols(const bf16* __restrict__ Bm, int ldn,
                                                  int n0, int k0) {
  int lane = threadIdx.x & 31;
  int half = lane >> 4, c = lane & 15;
  const bf16* p = Bm + (size_t)(k0 + half * 8) * ldn + n0 + c;
  union { v16bf v; bf16 e[16]; } u;
#pragma unroll
  for (int i = 0; i < 16; ++i) u.e[i] = p[(size_t)(i + (i & 8)) * ldn];
  return u.v;
}

// ---------------------------------------------------------------------------
// One-time weight preparation: f32 -> bf16 streams (halves weight HBM traffic
// for every GEMM; removes per-tile cvt from all inner loops)
// ---------------------------------------------------------------------------
__global__ __launch_bounds__(256) void cvt_bf16_kernel(const float* __restrict__ src,
                                                       bf16* __restrict__ dst) {
  size_t i = ((size_t)blockIdx.x * 256 + threadIdx.x) * 4;
  float4 v = *reinterpret_cast<const float4*>(src + i);
  dst[i + 0] = (bf16)v.x;
  dst[i + 1] = (bf16)v.y;
  dst[i + 2] = (bf16)v.z;
  dst[i + 3] = (bf16)v.w;
}

// W_U [D_MODEL, D_VOCAB] f32  ->  W_U^T [D_VOCAB, D_MODEL] bf16 (LDS-tiled)
__global__ __launch_bounds__(256) void transpose_cvt_kernel(const float* __restrict__ src,
                                                            bf16* __restrict__ dst,
                                                            int rows, int cols) {
  __shared__ float tile[32][33];
  int c0 = blockIdx.x * 32, r0 = blockIdx.y * 32;
  int tx = threadIdx.x & 31, ty = threadIdx.x >> 5;  // 8 rows per pass
#pragma unroll
  for (int i = 0; i < 32; i += 8)
    tile[ty + i][tx] = src[(size_t)(r0 + ty + i) * cols + c0 + tx];
  __syncthreads();
#pragma unroll
  for (int i = 0; i < 32; i += 8)
    dst[(size_t)(c0 + ty + i) * rows + r0 + tx] = (bf16)tile[tx][ty + i];
}

// ---------------------------------------------------------------------------
// Embedding: x = W_E.T[tok] + W_pos ; emit f32 residual + bf16 GEMM input
// ---------------------------------------------------------------------------
__global__ __launch_bounds__(256) void embed_kernel(
    const int* __restrict__ tokens, const float* __restrict__ W_E,
    const float* __restrict__ W_pos, float* __restrict__ xF, bf16* __restrict__ xB) {
  size_t i = (size_t)blockIdx.x * blockDim.x + threadIdx.x;
  int d   = (int)(i & (D_MODEL - 1));
  int bp  = (int)(i >> 10);
  int pos = bp & (SEQ - 1);
  int tok = tokens[bp];
  float val = W_E[(size_t)d * D_VOCAB + tok] + W_pos[(size_t)pos * D_MODEL + d];
  xF[i] = val;
  xB[i] = (bf16)val;
}

// ---------------------------------------------------------------------------
// Generic WMMA GEMM: C[M,N] = A[M,K](bf16) * W[N,K](bf16)^T
// wave tile 16x64 (A frag reused 4x), 8 waves/block -> 128x64 per block
// ---------------------------------------------------------------------------
enum { GM_QKV = 0, GM_RESID = 1, GM_RESID_BIAS = 2, GM_BIAS_RELU = 3, GM_LOGITS = 4 };

template <int MODE>
__global__ __launch_bounds__(256) void gemm_wmma_kernel(
    const bf16* __restrict__ A, const bf16* __restrict__ W,
    const float* __restrict__ bias, float* __restrict__ outF,
    bf16* __restrict__ outB, int M, int N, int K) {
  int wave = threadIdx.x >> 5;
  int lane = threadIdx.x & 31;
  int m0 = blockIdx.y * 128 + wave * 16;
  int n0 = blockIdx.x * 64;

  union Acc { v8f v; float f[8]; };
  Acc acc[4];
#pragma unroll
  for (int t = 0; t < 4; ++t)
#pragma unroll
    for (int r = 0; r < 8; ++r) acc[t].f[r] = 0.0f;

  for (int k0 = 0; k0 < K; k0 += 32) {
    if (k0 + 32 < K)  // hint next A chunk into cache (global_prefetch_b8)
      __builtin_prefetch(A + (size_t)(m0 + (lane & 15)) * K + k0 + 32, 0, 0);
    v16bf a = load_a_bf16(A, K, m0, k0);
#pragma unroll
    for (int t = 0; t < 4; ++t) {
      v16bf b = load_b_bf16_rows(W, K, n0 + t * 16, k0);
      acc[t].v = wmma_bf16(a, b, acc[t].v);
    }
  }

  int half = lane >> 4, c = lane & 15;
#pragma unroll
  for (int t = 0; t < 4; ++t) {
    int n = n0 + t * 16 + c;
    float bval = 0.0f;
    if constexpr (MODE == GM_RESID_BIAS || MODE == GM_BIAS_RELU) bval = bias[n];
#pragma unroll
    for (int r = 0; r < 8; ++r) {
      int m = m0 + r + 8 * half;   // C/D layout: row = vgpr + 8*(lane>=16)
      float v = acc[t].f[r] + bval;
      if constexpr (MODE == GM_RESID || MODE == GM_RESID_BIAS)
        v += outF[(size_t)m * N + n];
      if constexpr (MODE == GM_BIAS_RELU) v = fmaxf(v, 0.0f);
      if constexpr (MODE == GM_QKV) {
        int b_ = m >> 11, p_ = m & (SEQ - 1);     // m = b*SEQ + p
        int head = n >> 6, hd = n & (D_HEAD - 1); // n = head*64 + hd
        outB[(((size_t)b_ * N_HEADS + head) * SEQ + p_) * D_HEAD + hd] = (bf16)v;
      } else if constexpr (MODE == GM_LOGITS) {
        outF[(size_t)m * N + n] = v;
      } else if constexpr (MODE == GM_BIAS_RELU) {
        outB[(size_t)m * N + n] = (bf16)v;
      } else {  // residual modes: update f32 stream and its bf16 shadow
        outF[(size_t)m * N + n] = v;
        outB[(size_t)m * N + n] = (bf16)v;
      }
    }
  }
}

// ---------------------------------------------------------------------------
// Fused causal flash attention. Grid: (SEQ/128, BATCH*N_HEADS), 8 waves/block,
// each wave owns 16 query rows, streams 32-key blocks through WMMA + online
// softmax; P round-trips through per-wave LDS to reach A-fragment layout.
// ---------------------------------------------------------------------------
__global__ __launch_bounds__(256) void attention_kernel(
    const bf16* __restrict__ Q, const bf16* __restrict__ Kc,
    const bf16* __restrict__ V, bf16* __restrict__ Z) {
  __shared__ __align__(16) bf16 Plds[8][16 * 32];
  int wave = threadIdx.x >> 5;
  int lane = threadIdx.x & 31;
  int half = lane >> 4, l16 = lane & 15;
  int bh = blockIdx.y;
  int b = bh >> 4, head = bh & 15;
  int qbase = blockIdx.x * 128 + wave * 16;

  const bf16* q = Q  + (size_t)bh * SEQ * D_HEAD;
  const bf16* k = Kc + (size_t)bh * SEQ * D_HEAD;
  const bf16* v = V  + (size_t)bh * SEQ * D_HEAD;

  v16bf aq0 = load_a_bf16(q, D_HEAD, qbase, 0);
  v16bf aq1 = load_a_bf16(q, D_HEAD, qbase, 32);

  union Acc { v8f v; float f[8]; };
  Acc o[4];
  float mstate[8], lstate[8];
#pragma unroll
  for (int r = 0; r < 8; ++r) { mstate[r] = -3.0e38f; lstate[r] = 0.0f; }
#pragma unroll
  for (int t = 0; t < 4; ++t)
#pragma unroll
    for (int r = 0; r < 8; ++r) o[t].f[r] = 0.0f;

  int nblk = (qbase + 15) / 32 + 1;  // causal bound, uniform per wave
  for (int kb32 = 0; kb32 < nblk; ++kb32) {
    int kb = kb32 * 32;
    // scores S = q . k^T  (two 16x16 tiles covering 32 keys, K=64 in 2 steps)
    Acc c0, c1;
#pragma unroll
    for (int r = 0; r < 8; ++r) { c0.f[r] = 0.0f; c1.f[r] = 0.0f; }
    v16bf bk;
    bk = load_b_bf16_rows(k, D_HEAD, kb,       0);  c0.v = wmma_bf16(aq0, bk, c0.v);
    bk = load_b_bf16_rows(k, D_HEAD, kb,      32);  c0.v = wmma_bf16(aq1, bk, c0.v);
    bk = load_b_bf16_rows(k, D_HEAD, kb + 16,  0);  c1.v = wmma_bf16(aq0, bk, c1.v);
    bk = load_b_bf16_rows(k, D_HEAD, kb + 16, 32);  c1.v = wmma_bf16(aq1, bk, c1.v);

    // online softmax; a score row spans 16 lanes of one half-wave
#pragma unroll
    for (int r = 0; r < 8; ++r) {
      int qi = qbase + r + 8 * half;
      float s0 = (((kb + l16)      <= qi) ? c0.f[r] : -1.0e10f) * ATT_SCALE;
      float s1 = (((kb + 16 + l16) <= qi) ? c1.f[r] : -1.0e10f) * ATT_SCALE;
      float mx = fmaxf(s0, s1);
#pragma unroll
      for (int off = 8; off > 0; off >>= 1) mx = fmaxf(mx, __shfl_xor(mx, off, 16));
      float mnew = fmaxf(mstate[r], mx);
      float p0 = __expf(s0 - mnew), p1 = __expf(s1 - mnew);
      float psum = p0 + p1;
#pragma unroll
      for (int off = 8; off > 0; off >>= 1) psum += __shfl_xor(psum, off, 16);
      float alpha = __expf(mstate[r] - mnew);
      mstate[r] = mnew;
      lstate[r] = lstate[r] * alpha + psum;
#pragma unroll
      for (int t = 0; t < 4; ++t) o[t].f[r] *= alpha;
      int mrow = r + 8 * half;
      Plds[wave][mrow * 32 + l16]      = (bf16)p0;
      Plds[wave][mrow * 32 + 16 + l16] = (bf16)p1;
    }
    // same-wave DS ops are in-order; wait for LDS writes before re-reading
    asm volatile("s_wait_dscnt 0" ::: "memory");
    v16bf ap = load_a_bf16(&Plds[wave][0], 32, 0, 0);
#pragma unroll
    for (int t = 0; t < 4; ++t) {
      v16bf bv = load_b_bf16_cols(v, D_HEAD, t * 16, kb);  // V tile 32x64
      o[t].v = wmma_bf16(ap, bv, o[t].v);
    }
  }

  // epilogue: O /= l ; write z_flat[b, q, head*64 + h] as bf16
#pragma unroll
  for (int r = 0; r < 8; ++r) {
    int qi = qbase + r + 8 * half;
    float inv = 1.0f / lstate[r];
    size_t base = ((size_t)b * SEQ + qi) * (N_HEADS * D_HEAD) + head * D_HEAD;
#pragma unroll
    for (int t = 0; t < 4; ++t)
      Z[base + t * 16 + l16] = (bf16)(o[t].f[r] * inv);
  }
}

// ---------------------------------------------------------------------------
// Host orchestration
// ---------------------------------------------------------------------------
extern "C" void kernel_launch(void* const* d_in, const int* in_sizes, int n_in,
                              void* d_out, int out_size, void* d_ws, size_t ws_size,
                              hipStream_t stream) {
  const int*   tokens = (const int*)d_in[0];
  const float* W_E    = (const float*)d_in[1];
  const float* W_pos  = (const float*)d_in[2];
  const float* W_K    = (const float*)d_in[3];
  const float* W_Q    = (const float*)d_in[4];
  const float* W_V    = (const float*)d_in[5];
  const float* W_O    = (const float*)d_in[6];
  const float* W_in   = (const float*)d_in[7];
  const float* b_in   = (const float*)d_in[8];
  const float* W_out  = (const float*)d_in[9];
  const float* b_out  = (const float*)d_in[10];
  const float* W_U    = (const float*)d_in[11];
  float* logits = (float*)d_out;

  char* wp = (char*)d_ws;
  auto alloc = [&](size_t bytes) {
    char* r = wp;
    wp += (bytes + 255) & ~(size_t)255;
    return r;
  };
  // activations
  float* xF   = (float*)alloc((size_t)NTOK * D_MODEL * sizeof(float));
  bf16*  xB   = (bf16*) alloc((size_t)NTOK * D_MODEL * sizeof(bf16));
  bf16*  qB   = (bf16*) alloc((size_t)NTOK * D_MODEL * sizeof(bf16));
  bf16*  kB   = (bf16*) alloc((size_t)NTOK * D_MODEL * sizeof(bf16));
  bf16*  vB   = (bf16*) alloc((size_t)NTOK * D_MODEL * sizeof(bf16));
  bf16*  zB   = (bf16*) alloc((size_t)NTOK * D_MODEL * sizeof(bf16));
  bf16*  post = (bf16*) alloc((size_t)NTOK * D_MLP   * sizeof(bf16));
  // bf16 weight caches (all layers), converted once per launch
  const size_t nATT = (size_t)N_LAYERS * D_MODEL * D_MODEL;        // 4M each
  const size_t nMLP = (size_t)N_LAYERS * D_MLP * D_MODEL;          // 16M each
  const size_t nWU  = (size_t)D_VOCAB * D_MODEL;                   // 32.768M
  bf16* wqB = (bf16*)alloc(nATT * sizeof(bf16));
  bf16* wkB = (bf16*)alloc(nATT * sizeof(bf16));
  bf16* wvB = (bf16*)alloc(nATT * sizeof(bf16));
  bf16* woB = (bf16*)alloc(nATT * sizeof(bf16));
  bf16* wiB = (bf16*)alloc(nMLP * sizeof(bf16));
  bf16* wuoB = (bf16*)alloc(nMLP * sizeof(bf16));
  bf16* wutB = (bf16*)alloc(nWU * sizeof(bf16));   // W_U^T [D_VOCAB, D_MODEL]

  dim3 blk(256);
  dim3 g_dm(D_MODEL / 64, NTOK / 128);   // N=1024 GEMMs
  dim3 g_mlp(D_MLP / 64, NTOK / 128);    // N=4096 GEMM
  dim3 g_att(SEQ / 128, BATCHN * N_HEADS);
  dim3 g_out(D_VOCAB / 64, NTOK / 128);  // N=32000 GEMM

  // weight prep (one pass over all weights, f32 -> bf16)
  cvt_bf16_kernel<<<(unsigned)(nATT / 1024), blk, 0, stream>>>(W_Q, wqB);
  cvt_bf16_kernel<<<(unsigned)(nATT / 1024), blk, 0, stream>>>(W_K, wkB);
  cvt_bf16_kernel<<<(unsigned)(nATT / 1024), blk, 0, stream>>>(W_V, wvB);
  cvt_bf16_kernel<<<(unsigned)(nATT / 1024), blk, 0, stream>>>(W_O, woB);
  cvt_bf16_kernel<<<(unsigned)(nMLP / 1024), blk, 0, stream>>>(W_in, wiB);
  cvt_bf16_kernel<<<(unsigned)(nMLP / 1024), blk, 0, stream>>>(W_out, wuoB);
  transpose_cvt_kernel<<<dim3(D_VOCAB / 32, D_MODEL / 32), blk, 0, stream>>>(
      W_U, wutB, D_MODEL, D_VOCAB);

  embed_kernel<<<(NTOK * D_MODEL) / 256, blk, 0, stream>>>(tokens, W_E, W_pos, xF, xB);

  for (int l = 0; l < N_LAYERS; ++l) {
    const size_t offA = (size_t)l * D_MODEL * D_MODEL;
    const size_t offM = (size_t)l * D_MLP * D_MODEL;

    gemm_wmma_kernel<GM_QKV><<<g_dm, blk, 0, stream>>>(xB, wqB + offA, nullptr, nullptr,
                                                       qB, NTOK, D_MODEL, D_MODEL);
    gemm_wmma_kernel<GM_QKV><<<g_dm, blk, 0, stream>>>(xB, wkB + offA, nullptr, nullptr,
                                                       kB, NTOK, D_MODEL, D_MODEL);
    gemm_wmma_kernel<GM_QKV><<<g_dm, blk, 0, stream>>>(xB, wvB + offA, nullptr, nullptr,
                                                       vB, NTOK, D_MODEL, D_MODEL);

    attention_kernel<<<g_att, blk, 0, stream>>>(qB, kB, vB, zB);

    gemm_wmma_kernel<GM_RESID><<<g_dm, blk, 0, stream>>>(zB, woB + offA, nullptr, xF, xB,
                                                         NTOK, D_MODEL, D_MODEL);
    gemm_wmma_kernel<GM_BIAS_RELU><<<g_mlp, blk, 0, stream>>>(
        xB, wiB + offM, b_in + (size_t)l * D_MLP, nullptr, post, NTOK, D_MLP, D_MODEL);
    gemm_wmma_kernel<GM_RESID_BIAS><<<g_dm, blk, 0, stream>>>(
        post, wuoB + offM, b_out + (size_t)l * D_MODEL, xF, xB, NTOK, D_MODEL, D_MLP);
  }

  gemm_wmma_kernel<GM_LOGITS><<<g_out, blk, 0, stream>>>(xB, wutB, nullptr, logits,
                                                         nullptr, NTOK, D_VOCAB, D_MODEL);
}
